// CGCNNLayer_12575664242923
// MI455X (gfx1250) — compile-verified, matrix operand order checked by reading, not run on hardware
//
#include <hip/hip_runtime.h>
#include <math.h>

// ---------------------------------------------------------------------------
// CGCNN layer for MI455X (gfx1250, wave32, WMMA).
//   z[e,:] = W @ concat(node[src[e]], node[dst[e]], edge[e])      (GEMM, bf16 WMMA)
//   BN over E rows, msg = sigmoid(z[:, :128]) * softplus(z[:,128:])
//   out = node_feats; out[src[e]] += msg[e]                       (f32 atomics)
// Bias b cancels exactly under training-mode BN, so it is omitted.
// Memory-bound (~1.2 GB HBM @ 23.3 TB/s => ~50us); GEMM recomputed in pass 2
// instead of spilling a 614 MB z tensor. K loop = outer segment loop (3) x
// fully unrolled 4 K-steps: pointer selection hoisted, immediate offsets,
// VGPRs < 256 for occupancy.
// ---------------------------------------------------------------------------

typedef __attribute__((ext_vector_type(16))) __bf16 v16bf;
typedef __attribute__((ext_vector_type(8)))  __bf16 v8bf;
typedef __attribute__((ext_vector_type(8)))  float  v8f;

#define HD   128          // node feature dim
#define CDIM 256          // 2H output columns
#define KDIM 384          // 2H + ED input columns

// Pack one 16x32 bf16 A fragment (ISA layout: lanes 0-15 K{0..7,16..23},
// lanes 16-31 K{8..15,24..31}); p already includes khalf*8 and k offset.
__device__ __forceinline__ v16bf pack_afrag(const float* __restrict__ p, bool valid) {
    float4 f0 = ((const float4*)p)[0];
    float4 f1 = ((const float4*)p)[1];
    float4 f2 = ((const float4*)(p + 16))[0];
    float4 f3 = ((const float4*)(p + 16))[1];
    if (!valid) {
        f0 = make_float4(0.f, 0.f, 0.f, 0.f); f1 = f0; f2 = f0; f3 = f0;
    }
    float fa[16] = {f0.x, f0.y, f0.z, f0.w, f1.x, f1.y, f1.z, f1.w,
                    f2.x, f2.y, f2.z, f2.w, f3.x, f3.y, f3.z, f3.w};
    v16bf r;
#pragma unroll
    for (int i = 0; i < 16; ++i) r[i] = (__bf16)fa[i];
    return r;
}

// B is 32x16 (KxN): lane n (=lane&15) holds W column n, same K split as A.
// p already includes col*KDIM + khalf*8 + k offset.
__device__ __forceinline__ v16bf load_bfrag(const __bf16* __restrict__ p) {
    v8bf c0 = *(const v8bf*)p;
    v8bf c1 = *(const v8bf*)(p + 16);
    v16bf r;
#pragma unroll
    for (int i = 0; i < 8; ++i) { r[i] = c0[i]; r[i + 8] = c1[i]; }
    return r;
}

// PASS 0: GEMM + per-column sum / sum-of-squares accumulation.
// PASS 1: GEMM + BN affine + sigmoid*softplus + scatter-add into out.
template <int PASS>
__global__ __launch_bounds__(256) void gemm_kernel(
    const float* __restrict__ node_feats,
    const float* __restrict__ edge_feats,
    const int*   __restrict__ srcIdx,
    const int*   __restrict__ dstIdx,
    const __bf16* __restrict__ Wb,          // [256][384] bf16
    float* __restrict__ statSum,            // [256]
    float* __restrict__ statSqs,            // [256]
    const float* __restrict__ bnScale,      // [256]
    const float* __restrict__ bnShift,      // [256]
    float* __restrict__ out,                // [N][128]
    int E, int numTiles)                    // numTiles = ceil(E/32)
{
    const int lane  = threadIdx.x & 31;
    const int wave  = threadIdx.x >> 5;     // 0..7, one wave per (gate,value) pair
    const int n     = lane & 15;
    const int khalf = lane >> 4;

    const int tileA = wave;                 // gate columns  [wave*16 .. +16)
    const int tileB = wave + 8;             // value columns [128 + wave*16 .. +16)

    // tile-invariant B fragment base addresses
    const __bf16* wpA = Wb + (size_t)(tileA * 16 + n) * KDIM + khalf * 8;
    const __bf16* wpB = Wb + (size_t)(tileB * 16 + n) * KDIM + khalf * 8;

    // vector stat accumulators (horizontal reduce deferred to kernel end)
    v8f sumA = {0.f,0.f,0.f,0.f,0.f,0.f,0.f,0.f};
    v8f sqA  = sumA, sumB = sumA, sqB = sumA;

    for (int t = blockIdx.x; t < numTiles; t += gridDim.x) {
        const int base = t * 32;            // 32 edges per block iteration

        // prefetch next iteration's streamed edge_feats rows (global_prefetch_b8)
        {
            const long nb = (long)(t + gridDim.x) * 32;
            if (nb + lane < E) {
                const char* np = (const char*)(edge_feats + (size_t)(nb + lane) * HD);
#pragma unroll
                for (int c = 0; c < 4; ++c) __builtin_prefetch(np + c * 128, 0, 1);
            }
        }

        const int  r0 = base + n,      r1 = base + 16 + n;
        const bool va0 = (r0 < E),     va1 = (r1 < E);
        const int  e0 = va0 ? r0 : 0,  e1 = va1 ? r1 : 0;
        const int  si0 = srcIdx[e0],   di0 = dstIdx[e0];
        const int  si1 = srcIdx[e1],   di1 = dstIdx[e1];

        v8f acc00 = {0.f,0.f,0.f,0.f,0.f,0.f,0.f,0.f};  // edges[0:16) x gate
        v8f acc01 = acc00;                               // edges[0:16) x value
        v8f acc10 = acc00;                               // edges[16:32) x gate
        v8f acc11 = acc00;                               // edges[16:32) x value

#pragma unroll 1
        for (int seg = 0; seg < 3; ++seg) {              // src | dst | edge
            const float* p0;
            const float* p1;
            if (seg == 0)      { p0 = node_feats + (size_t)si0 * HD;
                                 p1 = node_feats + (size_t)si1 * HD; }
            else if (seg == 1) { p0 = node_feats + (size_t)di0 * HD;
                                 p1 = node_feats + (size_t)di1 * HD; }
            else               { p0 = edge_feats + (size_t)e0 * HD;
                                 p1 = edge_feats + (size_t)e1 * HD; }
            p0 += khalf * 8;
            p1 += khalf * 8;
            const __bf16* wA = wpA + seg * 128;
            const __bf16* wB = wpB + seg * 128;

#pragma unroll
            for (int q = 0; q < 4; ++q) {                // 4 K-steps of 32
                const int off = q * 32;
                v16bf a0 = pack_afrag(p0 + off, va0);
                v16bf a1 = pack_afrag(p1 + off, va1);
                v16bf bA = load_bfrag(wA + off);
                v16bf bB = load_bfrag(wB + off);

                acc00 = __builtin_amdgcn_wmma_f32_16x16x32_bf16(
                    false, a0, false, bA, (short)0, acc00, false, false);
                acc10 = __builtin_amdgcn_wmma_f32_16x16x32_bf16(
                    false, a1, false, bA, (short)0, acc10, false, false);
                acc01 = __builtin_amdgcn_wmma_f32_16x16x32_bf16(
                    false, a0, false, bB, (short)0, acc01, false, false);
                acc11 = __builtin_amdgcn_wmma_f32_16x16x32_bf16(
                    false, a1, false, bB, (short)0, acc11, false, false);
            }
        }

        if (PASS == 0) {
            sumA += acc00 + acc10;
            sqA  += acc00 * acc00 + acc10 * acc10;
            sumB += acc01 + acc11;
            sqB  += acc01 * acc01 + acc11 * acc11;
        } else {
            const int   colA = tileA * 16 + n;       // 0..127 (gate)
            const int   colB = tileB * 16 + n;       // 128..255 (value)
            const float scA = bnScale[colA], shA = bnShift[colA];
            const float scB = bnScale[colB], shB = bnShift[colB];
#pragma unroll
            for (int half = 0; half < 2; ++half) {
                const v8f& ag = half ? acc10 : acc00;
                const v8f& av = half ? acc11 : acc01;
#pragma unroll
                for (int v = 0; v < 8; ++v) {
                    const int m = v + 8 * khalf;     // C/D layout: M = v + 8*(lane>>4)
                    const int e = base + half * 16 + m;
                    if (e < E) {
                        float g   = ag[v] * scA + shA;
                        float x   = av[v] * scB + shB;
                        float sig = 1.0f / (1.0f + __expf(-g));
                        float sp  = (x > 20.0f) ? x : log1pf(__expf(x));
                        float msg = sig * sp;
                        int   sm  = srcIdx[e];
                        atomicAdd(out + (size_t)sm * HD + colA, msg);
                    }
                }
            }
        }
    }

    if (PASS == 0) {
        float sA = 0.f, qA = 0.f, sB = 0.f, qB = 0.f;
#pragma unroll
        for (int v = 0; v < 8; ++v) {
            sA += sumA[v]; qA += sqA[v];
            sB += sumB[v]; qB += sqB[v];
        }
        // lanes l and l+16 hold the same column N=l&15 -> pair reduce
        sA += __shfl_xor(sA, 16, 32); qA += __shfl_xor(qA, 16, 32);
        sB += __shfl_xor(sB, 16, 32); qB += __shfl_xor(qB, 16, 32);
        if (lane < 16) {
            atomicAdd(statSum + tileA * 16 + n, sA);
            atomicAdd(statSqs + tileA * 16 + n, qA);
            atomicAdd(statSum + tileB * 16 + n, sB);
            atomicAdd(statSqs + tileB * 16 + n, qB);
        }
    }
}

__global__ void prep_kernel(const float* __restrict__ W, __bf16* __restrict__ Wb,
                            float* __restrict__ stats, int nW) {
    int i = blockIdx.x * blockDim.x + threadIdx.x;
    if (i < nW)       Wb[i] = (__bf16)W[i];
    if (i < 2 * CDIM) stats[i] = 0.0f;
}

__global__ void bn_finalize_kernel(const float* __restrict__ stats,
                                   const float* __restrict__ gamma,
                                   const float* __restrict__ beta,
                                   float* __restrict__ scsh, float invE) {
    int c = threadIdx.x;                 // 256 threads
    float mean = stats[c] * invE;
    float var  = stats[CDIM + c] * invE - mean * mean;
    float inv  = rsqrtf(var + 1e-5f);
    float sc   = gamma[c] * inv;
    scsh[c]        = sc;
    scsh[CDIM + c] = beta[c] - mean * sc;
}

extern "C" void kernel_launch(void* const* d_in, const int* in_sizes, int n_in,
                              void* d_out, int out_size, void* d_ws, size_t ws_size,
                              hipStream_t stream) {
    const float* node_feats = (const float*)d_in[0];
    const float* edge_feats = (const float*)d_in[1];
    const float* W          = (const float*)d_in[2];
    // d_in[3] = b : exactly cancelled by batch-norm mean subtraction
    const float* gamma      = (const float*)d_in[4];
    const float* beta       = (const float*)d_in[5];
    const int*   eidx       = (const int*)d_in[6];   // [2,E]

    const int N = in_sizes[0] / HD;
    const int E = in_sizes[1] / HD;
    const int* src = eidx;
    const int* dst = eidx + E;

    // workspace layout
    char*   ws    = (char*)d_ws;
    __bf16* Wb    = (__bf16*)ws;                         // 256*384*2 = 196608 B
    float*  stats = (float*)(ws + 196608);               // sum[256] | sumsq[256]
    float*  scsh  = (float*)(ws + 196608 + 2048);        // scale[256] | shift[256]
    float*  out   = (float*)d_out;

    // residual base: out = node_feats
    hipMemcpyAsync(out, node_feats, (size_t)N * HD * sizeof(float),
                   hipMemcpyDeviceToDevice, stream);

    const int nW = CDIM * KDIM;                          // 98304
    prep_kernel<<<(nW + 255) / 256, 256, 0, stream>>>(W, Wb, stats, nW);

    const int numTiles = (E + 31) / 32;
    int grid = numTiles < 4096 ? numTiles : 4096;

    gemm_kernel<0><<<grid, 256, 0, stream>>>(node_feats, edge_feats, src, dst, Wb,
                                             stats, stats + CDIM, nullptr, nullptr,
                                             nullptr, E, numTiles);

    bn_finalize_kernel<<<1, 256, 0, stream>>>(stats, gamma, beta, scsh,
                                              1.0f / (float)E);

    gemm_kernel<1><<<grid, 256, 0, stream>>>(node_feats, edge_feats, src, dst, Wb,
                                             stats, stats + CDIM, scsh, scsh + CDIM,
                                             out, E, numTiles);
}
